// MyModel_87522843559104
// MI455X (gfx1250) — matrix operand order repaired and checked by reading, not compiled, not run on hardware
//
#include <hip/hip_runtime.h>
#include <hip/hip_bf16.h>

typedef __attribute__((ext_vector_type(16))) _Float16 v16h;
typedef __attribute__((ext_vector_type(8)))  _Float16 v8h;
typedef __attribute__((ext_vector_type(8)))  float    v8f;

#define CAT16(lo, hi) __builtin_shufflevector(lo, hi, 0,1,2,3,4,5,6,7,8,9,10,11,12,13,14,15)

// ---------------------------------------------------------------------------
// f32 -> f16 elementwise convert (vector-4 per thread)
__global__ __launch_bounds__(256)
void cvt_f16_kernel(const float* __restrict__ in, _Float16* __restrict__ out, int n)
{
    int i = (blockIdx.x * 256 + threadIdx.x) * 4;
    if (i + 3 < n) {
        float4 v = *(const float4*)(in + i);
        out[i + 0] = (_Float16)v.x;
        out[i + 1] = (_Float16)v.y;
        out[i + 2] = (_Float16)v.z;
        out[i + 3] = (_Float16)v.w;
    }
}

// W (K x N) f32 -> Wt (N x K) f16.  grid = (K/256, N); no div/mod.
__global__ __launch_bounds__(256)
void transpose_f16_kernel(const float* __restrict__ in, _Float16* __restrict__ out,
                          int K, int N)
{
    int k = blockIdx.x * 256 + threadIdx.x;
    int n = blockIdx.y;
    out[(long long)n * K + k] = (_Float16)in[(long long)k * N + n];
}

// ---------------------------------------------------------------------------
// LDS-free WMMA GEMM:  C = A(MxK,f16) * Bt(NxK,f16)^T + bias
// Fragments stream straight from global as 128-bit loads (L0/L2 reuse).
// Block 256 thr = 8 waves (2 M x 4 N); wave tile 32x64 => 8 WMMAs / K-chunk.
// Requires M % 64 == 0, K % 32 == 0. NGUARD=true handles ragged N (col-guarded
// stores; OOB B pointers clamped to row 0 so the hot loop stays branch-free).
template <bool NGUARD>
__global__ __launch_bounds__(256)
void gemm_wmma_f16(const _Float16* __restrict__ A,   // M x K
                   const _Float16* __restrict__ Bt,  // N x K (pre-transposed)
                   const float*    __restrict__ bias,
                   float*          __restrict__ Cf32, // nullable
                   _Float16*       __restrict__ Cf16, // nullable
                   int M, int N, int K, int ldc, long long c_base, int relu)
{
    const int tid  = threadIdx.x;
    const int lane = tid & 31;
    const int wave = tid >> 5;
    const int wm   = wave >> 2;               // 0..1
    const int wn   = wave & 3;                // 0..3
    const int m0   = blockIdx.x * 64 + wm * 32;
    const int n0   = blockIdx.y * 256 + wn * 64;
    const int l15  = lane & 15;
    const int kh   = lane >> 4;               // K-half select per ISA layout

    v8f acc[2][4];
    #pragma unroll
    for (int i = 0; i < 2; ++i)
        #pragma unroll
        for (int s = 0; s < 4; ++s) acc[i][s] = (v8f){};

    const _Float16* Ar0 = A + (long long)(m0 + l15) * K;
    const _Float16* Ar1 = A + (long long)(m0 + 16 + l15) * K;
    const _Float16* Bc[4];
    #pragma unroll
    for (int s = 0; s < 4; ++s) {
        int col = n0 + s * 16 + l15;
        if (NGUARD && col >= N) col = 0;      // clamp: loads valid, result unused
        Bc[s] = Bt + (long long)col * K;
    }

    for (int kk = 0; kk < K; kk += 32) {
        __builtin_prefetch(Ar0 + kk + 32, 0, 3);   // speculative; overshoot is dropped

        // A fragments (16x32 f16, ISA layout: two 16B chunks per lane)
        v16h af0 = CAT16(*(const v8h*)(Ar0 + kk + 8 * kh),
                         *(const v8h*)(Ar0 + kk + 16 + 8 * kh));
        v16h af1 = CAT16(*(const v8h*)(Ar1 + kk + 8 * kh),
                         *(const v8h*)(Ar1 + kk + 16 + 8 * kh));

        #pragma unroll
        for (int s = 0; s < 4; ++s) {
            v16h bf = CAT16(*(const v8h*)(Bc[s] + kk + 8 * kh),
                            *(const v8h*)(Bc[s] + kk + 16 + 8 * kh));
            acc[0][s] = __builtin_amdgcn_wmma_f32_16x16x32_f16(
                false, af0, false, bf, (short)0, acc[0][s], false, false);
            acc[1][s] = __builtin_amdgcn_wmma_f32_16x16x32_f16(
                false, af1, false, bf, (short)0, acc[1][s], false, false);
        }
    }

    // Epilogue: D f32 layout — VGPR r: M = r + 8*kh; N = lane&15
    #pragma unroll
    for (int i = 0; i < 2; ++i) {
        #pragma unroll
        for (int s = 0; s < 4; ++s) {
            int col = n0 + s * 16 + l15;
            if (NGUARD && col >= N) continue;
            float bv = bias ? bias[col] : 0.0f;
            #pragma unroll
            for (int r = 0; r < 8; ++r) {
                int row = m0 + i * 16 + r + 8 * kh;
                float v = acc[i][s][r] + bv;
                if (relu) v = fmaxf(v, 0.0f);
                long long idx = c_base + (long long)row * ldc + col;
                if (Cf32) Cf32[idx] = v;
                if (Cf16) Cf16[idx] = (_Float16)v;
            }
        }
    }
}

// ---------------------------------------------------------------------------
// Fused Bahdanau attention + softmax + context + token-embed concat -> x (f16)
__global__ __launch_bounds__(256)
void attn_context_kernel(const float* __restrict__ feat_w1,   // (B, 64, 512)
                         const float* __restrict__ hid_w2,    // (B, 512) incl b2
                         const float* __restrict__ Vw,        // (512)
                         const float* __restrict__ bV,        // (1)
                         const float* __restrict__ features,  // (B, 64, 256) f32
                         const float* __restrict__ emb,       // (VOCAB, 256)
                         const int*   __restrict__ target,    // (B, 48)
                         int step,
                         _Float16* __restrict__ x)            // (B, 512) f16
{
    const int b    = blockIdx.x;
    const int tid  = threadIdx.x;
    const int lane = tid & 31;
    const int wave = tid >> 5;

    __shared__ float sc[64];
    __shared__ float ssum;

    const float* fw = feat_w1 + (long long)b * 64 * 512;
    const float* hw = hid_w2  + (long long)b * 512;

    for (int l = wave; l < 64; l += 8) {
        float s = 0.0f;
        const float* fwl = fw + l * 512;
        for (int u = lane; u < 512; u += 32)
            s += tanhf(fwl[u] + hw[u]) * Vw[u];
        #pragma unroll
        for (int off = 16; off > 0; off >>= 1)
            s += __shfl_down(s, off);
        if (lane == 0) sc[l] = s + bV[0];
    }
    __syncthreads();

    if (tid == 0) {
        float m = sc[0];
        #pragma unroll
        for (int l = 1; l < 64; ++l) m = fmaxf(m, sc[l]);
        float acc = 0.0f;
        #pragma unroll
        for (int l = 0; l < 64; ++l) { float e = __expf(sc[l] - m); sc[l] = e; acc += e; }
        ssum = acc;
    }
    __syncthreads();

    const float inv = 1.0f / ssum;
    const float* fb = features + (long long)b * 64 * 256;
    const int e = tid;                         // 256 threads <-> 256 channels
    float ctx = 0.0f;
    #pragma unroll 8
    for (int l = 0; l < 64; ++l)
        ctx += sc[l] * fb[l * 256 + e];
    x[(long long)b * 512 + e] = (_Float16)(ctx * inv);

    const int tok = (step == 0) ? 1 : target[b * 48 + step];
    x[(long long)b * 512 + 256 + e] = (_Float16)emb[(long long)tok * 256 + e];
}

// GRU gates with h0 == 0: mh == gru_b[1]; state = (1-z) * hh. Writes f16.
__global__ __launch_bounds__(256)
void gru_gates_kernel(const float* __restrict__ mx,     // (128, 1536) incl gru_b[0]
                      const float* __restrict__ gru_b,  // (2, 1536)
                      _Float16* __restrict__ hidden)    // (128, 512) f16
{
    int i = blockIdx.x * blockDim.x + threadIdx.x;
    if (i >= 128 * 512) return;
    int b = i >> 9;
    int u = i & 511;
    const float* gb1 = gru_b + 1536;
    const float* m   = mx + (long long)b * 1536;
    float z  = 1.0f / (1.0f + __expf(-(m[u]       + gb1[u])));
    float r  = 1.0f / (1.0f + __expf(-(m[512 + u] + gb1[512 + u])));
    float hh = tanhf(m[1024 + u] + r * gb1[1024 + u]);
    hidden[i] = (_Float16)((1.0f - z) * hh);
}

// ---------------------------------------------------------------------------
extern "C" void kernel_launch(void* const* d_in, const int* in_sizes, int n_in,
                              void* d_out, int out_size, void* d_ws, size_t ws_size,
                              hipStream_t stream) {
    const float* img    = (const float*)d_in[0];
    const int*   target = (const int*)  d_in[1];
    const float* W_fc   = (const float*)d_in[2];
    const float* b_fc   = (const float*)d_in[3];
    const float* W1     = (const float*)d_in[4];
    const float* b1     = (const float*)d_in[5];
    const float* W2     = (const float*)d_in[6];
    const float* b2     = (const float*)d_in[7];
    const float* Vw     = (const float*)d_in[8];
    const float* bV     = (const float*)d_in[9];
    const float* emb    = (const float*)d_in[10];
    const float* gru_k  = (const float*)d_in[11];
    // d_in[12] = gru_rk: dead (h0 == 0)
    const float* gru_b  = (const float*)d_in[13];
    const float* fc1_w  = (const float*)d_in[14];
    const float* fc1_b  = (const float*)d_in[15];
    const float* fc2_w  = (const float*)d_in[16];
    const float* fc2_b  = (const float*)d_in[17];
    float* out = (float*)d_out;

    // workspace layout (bytes, 256-aligned)
    char* ws = (char*)d_ws;
    size_t off = 0;
    auto alloc = [&](size_t bytes) { char* p = ws + off; off = (off + bytes + 255) & ~(size_t)255; return p; };
    _Float16* img_h    = (_Float16*)alloc(16777216ull * 2);   // 8192 x 2048
    _Float16* WfcT     = (_Float16*)alloc(2048 * 256 * 2);    // 256  x 2048
    _Float16* W1T      = (_Float16*)alloc(256 * 512 * 2);     // 512  x 256
    _Float16* W2T      = (_Float16*)alloc(512 * 512 * 2);     // 512  x 512
    _Float16* grukT    = (_Float16*)alloc(512 * 1536 * 2);    // 1536 x 512
    _Float16* fc1T     = (_Float16*)alloc(512 * 512 * 2);     // 512  x 512
    _Float16* fc2T     = (_Float16*)alloc(512 * 5000 * 2);    // 5000 x 512
    float*    features = (float*)   alloc(8192 * 256 * 4);
    _Float16* feat_h   = (_Float16*)alloc(8192 * 256 * 2);
    float*    feat_w1  = (float*)   alloc(8192 * 512 * 4);
    _Float16* hidden_h = (_Float16*)alloc(128 * 512 * 2);
    float*    hid_w2   = (float*)   alloc(128 * 512 * 4);
    _Float16* x_h      = (_Float16*)alloc(128 * 512 * 2);
    float*    mx       = (float*)   alloc(128 * 1536 * 4);
    _Float16* out1_h   = (_Float16*)alloc(128 * 512 * 2);

    dim3 blk(256);

    // --- one-time data layout pass: f16 activations + transposed f16 weights
    cvt_f16_kernel<<<16777216 / 1024, blk, 0, stream>>>(img, img_h, 16777216);
    transpose_f16_kernel<<<dim3(2048 / 256, 256),  blk, 0, stream>>>(W_fc,  WfcT, 2048, 256);
    transpose_f16_kernel<<<dim3(256 / 256, 512),   blk, 0, stream>>>(W1,    W1T,  256,  512);
    transpose_f16_kernel<<<dim3(512 / 256, 512),   blk, 0, stream>>>(W2,    W2T,  512,  512);
    transpose_f16_kernel<<<dim3(512 / 256, 1536),  blk, 0, stream>>>(gru_k, grukT,512, 1536);
    transpose_f16_kernel<<<dim3(512 / 256, 512),   blk, 0, stream>>>(fc1_w, fc1T, 512,  512);
    transpose_f16_kernel<<<dim3(512 / 256, 5000),  blk, 0, stream>>>(fc2_w, fc2T, 512, 5000);
    hipMemsetAsync(hidden_h, 0, 128 * 512 * sizeof(_Float16), stream);  // f16 zeros

    // --- encoder: features = relu(img @ W_fc + b_fc)   (8192 x 256, K=2048)
    gemm_wmma_f16<false><<<dim3(8192 / 64, 1), blk, 0, stream>>>(
        img_h, WfcT, b_fc, features, feat_h, 8192, 256, 2048, 256, 0, 1);
    // --- feat_w1 = features @ W1 + b1                   (8192 x 512, K=256)
    gemm_wmma_f16<false><<<dim3(8192 / 64, 2), blk, 0, stream>>>(
        feat_h, W1T, b1, feat_w1, nullptr, 8192, 512, 256, 512, 0, 0);

    const int T1 = 47;
    for (int t = 0; t < T1; ++t) {
        // hid_w2 = hidden @ W2 + b2                      (128 x 512, K=512)
        gemm_wmma_f16<false><<<dim3(2, 2), blk, 0, stream>>>(
            hidden_h, W2T, b2, hid_w2, nullptr, 128, 512, 512, 512, 0, 0);
        // attention + context + [context, emb[tok]] -> x (f16)
        attn_context_kernel<<<128, 256, 0, stream>>>(
            feat_w1, hid_w2, Vw, bV, features, emb, target, t, x_h);
        // mx = x @ gru_k + gru_b[0]                      (128 x 1536, K=512)
        gemm_wmma_f16<false><<<dim3(2, 6), blk, 0, stream>>>(
            x_h, grukT, gru_b, mx, nullptr, 128, 1536, 512, 1536, 0, 0);
        // gates -> new hidden/state (f16)
        gru_gates_kernel<<<(128 * 512) / 256, blk, 0, stream>>>(mx, gru_b, hidden_h);
        // out1 = state @ fc1_w + fc1_b                   (128 x 512, K=512) -> f16
        gemm_wmma_f16<false><<<dim3(2, 2), blk, 0, stream>>>(
            hidden_h, fc1T, fc1_b, nullptr, out1_h, 128, 512, 512, 512, 0, 0);
        // logits -> d_out[:, t, :]                       (128 x 5000, K=512) ragged N
        gemm_wmma_f16<true><<<dim3(2, (5000 + 255) / 256), blk, 0, stream>>>(
            out1_h, fc2T, fc2_b, out, nullptr, 128, 5000, 512, T1 * 5000,
            (long long)t * 5000, 0);
    }
}
